// SWAN_87591563034971
// MI455X (gfx1250) — compile-verified, matrix operand order checked by reading, not run on hardware
//
#include <hip/hip_runtime.h>
#include <math.h>

#define N_NODES 50000
#define N_EDGES 800000
#define C 64
#define IN_F 32
#define OUT_F 16
#define N_ITERS 3
#define GAMMA_C 0.1f
#define EPS_C 0.1f
#define BETA_C 0.1f
#define BN_EPS_C 1e-5f

typedef __attribute__((ext_vector_type(16))) __bf16 v16bf;
typedef __attribute__((ext_vector_type(8)))  __bf16 v8bf;
typedef __attribute__((ext_vector_type(8)))  float  v8f;

// ---------- helpers ----------

__device__ __forceinline__ unsigned short f2bf(float f) {
  unsigned int u = __float_as_uint(f);
  unsigned int r = (u + 0x7FFFu + ((u >> 16) & 1u)) >> 16;  // RNE
  return (unsigned short)r;
}

// A/B fragment K index for 16-bit WMMA (ISA 7.12.2, 16-bit A 16x32):
// element j in [0,8)  -> k = khalf*8 + j        (VGPR0-3)
// element j in [8,16) -> k = 16 + khalf*8 + j-8 (VGPR4-7)
__device__ __forceinline__ int frag_k(int j, int khalf) {
  int vg = j >> 1;
  return ((vg & 4) << 2) + (khalf << 3) + ((vg & 3) << 1) + (j & 1);
}

// Fragment = two contiguous 8-element bf16 runs -> two b128 loads.
// p points at bf16 element (row_base + kc*32 + khalf*8).
__device__ __forceinline__ v16bf load_frag_pair(const unsigned short* __restrict__ p) {
  union { v8bf h[2]; v16bf v; } f;
  f.h[0] = *(const v8bf*)(p);
  f.h[1] = *(const v8bf*)(p + 16);
  return f.v;
}

// B fragment for D = A * W^T : B[k][n] = W[n][k]; w row-major [outC][K] bf16
__device__ __forceinline__ v16bf load_bfragT(const unsigned short* __restrict__ w,
                                             int K, int nbase, int kc) {
  const int lane = threadIdx.x & 31;
  const int n = nbase + (lane & 15);
  const int khalf = lane >> 4;
  return load_frag_pair(w + (size_t)n * K + kc * 32 + (khalf << 3));
}

// ---------- small utility kernels ----------

__global__ void zero_kernel(float* __restrict__ p, size_t n) {
  size_t i = (size_t)blockIdx.x * blockDim.x + threadIdx.x;
  if (i < n) p[i] = 0.f;
}

__global__ void prep_weights_kernel(const float* __restrict__ W,
                                    const float* __restrict__ conv_w_raw,
                                    const float* __restrict__ sym_w_raw,
                                    const float* __restrict__ conv_we,
                                    const float* __restrict__ sym_we,
                                    const float* __restrict__ att_w2,
                                    const float* __restrict__ att_w3,
                                    unsigned short* __restrict__ w2bf,
                                    unsigned short* __restrict__ w3bf,
                                    unsigned short* __restrict__ cwebf,
                                    unsigned short* __restrict__ swebf,
                                    unsigned short* __restrict__ Wabf,
                                    unsigned short* __restrict__ Wsbf,
                                    unsigned short* __restrict__ AWbf) {
  const int t = threadIdx.x;
  for (int idx = t; idx < C * 2 * C; idx += blockDim.x)
    w2bf[idx] = f2bf(att_w2[idx]);
  for (int idx = t; idx < C * C; idx += blockDim.x) {
    w3bf[idx]  = f2bf(att_w3[idx]);
    cwebf[idx] = f2bf(conv_we[idx]);
    swebf[idx] = f2bf(sym_we[idx]);
    const int r = idx / C, c = idx % C;
    // Wa = triu(conv_w_raw,1) - triu(conv_w_raw,1)^T  (antisymmetric)
    float wa = (c > r ? conv_w_raw[idx] : 0.f) - (r > c ? conv_w_raw[c * C + r] : 0.f);
    // Ws = triu(sym_w_raw) + triu(sym_w_raw)^T        (symmetric, diag doubled)
    float wsy = (c >= r ? sym_w_raw[idx] : 0.f) + (r >= c ? sym_w_raw[c * C + r] : 0.f);
    // AW = W - W^T - GAMMA*I
    float aw = W[idx] - W[c * C + r] - ((r == c) ? GAMMA_C : 0.f);
    Wabf[idx] = f2bf(wa);
    Wsbf[idx] = f2bf(wsy);
    AWbf[idx] = f2bf(aw);
  }
}

__global__ void emb_kernel(const float* __restrict__ x_in,
                           const float* __restrict__ emb_w,
                           const float* __restrict__ emb_b,
                           float* __restrict__ x0,
                           unsigned short* __restrict__ x0bf) {
  int i = blockIdx.x * blockDim.x + threadIdx.x;
  if (i >= N_NODES) return;
  float xr[IN_F];
#pragma unroll
  for (int k = 0; k < IN_F; ++k) xr[k] = x_in[(size_t)i * IN_F + k];
#pragma unroll 4
  for (int o = 0; o < C; ++o) {
    float a = emb_b[o];
#pragma unroll
    for (int k = 0; k < IN_F; ++k) a += emb_w[o * IN_F + k] * xr[k];
    x0[(size_t)i * C + o]   = a;
    x0bf[(size_t)i * C + o] = f2bf(a);
  }
}

// ---------- edge MLP stage 1: h = [x[src],x[dst]] @ att_w2^T + b2 ; BN stats ----------

__global__ __launch_bounds__(32)
void mlp1_kernel(const unsigned short* __restrict__ x0bf,
                 const int* __restrict__ src, const int* __restrict__ dst,
                 const unsigned short* __restrict__ w2bf,  // [64][128] bf16
                 const float* __restrict__ b2,
                 float* __restrict__ h,
                 float* __restrict__ stat_sum, float* __restrict__ stat_sqs) {
  const int e0 = blockIdx.x * 16;
  const int lane = threadIdx.x & 31;
  const int m = lane & 15;
  const int khalf = lane >> 4;
  const int e = e0 + m;
  const int s = src[e], d = dst[e];
  const unsigned short* __restrict__ rs = x0bf + (size_t)s * C;
  const unsigned short* __restrict__ rd = x0bf + (size_t)d * C;

  // K = 128 in 4 chunks of 32; chunks 0,1 come from x[src], chunks 2,3 from x[dst]
  v16bf afr[4];
#pragma unroll
  for (int kc = 0; kc < 4; ++kc) {
    const unsigned short* row = (kc < 2) ? rs : (rd - C);
    afr[kc] = load_frag_pair(row + kc * 32 + (khalf << 3));
  }

  const int n_lane = lane & 15;
#pragma unroll
  for (int nt = 0; nt < 4; ++nt) {
    v8f acc = {0.f, 0.f, 0.f, 0.f, 0.f, 0.f, 0.f, 0.f};
#pragma unroll
    for (int kc = 0; kc < 4; ++kc) {
      v16bf b = load_bfragT(w2bf, 2 * C, nt * 16, kc);
      acc = __builtin_amdgcn_wmma_f32_16x16x32_bf16(false, afr[kc], false, b,
                                                    (short)0, acc, false, false);
    }
    const int n = nt * 16 + n_lane;
    const float bias = b2[n];
    float lsum = 0.f, lsqs = 0.f;
#pragma unroll
    for (int j = 0; j < 8; ++j) {
      const int mr = j + (khalf << 3);
      float v = acc[j] + bias;
      h[(size_t)(e0 + mr) * C + n] = v;
      lsum += v;
      lsqs += v * v;
    }
    atomicAdd(&stat_sum[n], lsum);
    atomicAdd(&stat_sqs[n], lsqs);
  }
}

__global__ void bn_finalize_kernel(const float* __restrict__ stat_sum,
                                   const float* __restrict__ stat_sqs,
                                   const float* __restrict__ bn_g,
                                   const float* __restrict__ bn_b,
                                   float* __restrict__ scale,
                                   float* __restrict__ shift) {
  int c = threadIdx.x;
  if (c < C) {
    float m = stat_sum[c] * (1.f / (float)N_EDGES);
    float v = stat_sqs[c] * (1.f / (float)N_EDGES) - m * m;
    float sc = bn_g[c] * rsqrtf(v + BN_EPS_C);
    scale[c] = sc;
    shift[c] = bn_b[c] - m * sc;
  }
}

// ---------- edge MLP stage 2: ew = relu(lrelu(BN(h),0.2) @ att_w3^T + b3); deg scatter ----------

__global__ __launch_bounds__(32)
void mlp2_kernel(const float* __restrict__ h,
                 const int* __restrict__ src,
                 const unsigned short* __restrict__ w3bf,  // [64][64] bf16
                 const float* __restrict__ b3,
                 const float* __restrict__ scale, const float* __restrict__ shift,
                 float* __restrict__ ew, float* __restrict__ deg) {
  const int e0 = blockIdx.x * 16;
  const int lane = threadIdx.x & 31;
  const int m = lane & 15;
  const int khalf = lane >> 4;
  const float* __restrict__ hr = h + (size_t)(e0 + m) * C;

  v16bf afr[2];
#pragma unroll
  for (int kc = 0; kc < 2; ++kc) {
    union { unsigned short s16[16]; v16bf v; } f;
#pragma unroll
    for (int j = 0; j < 16; ++j) {
      int k = kc * 32 + frag_k(j, khalf);
      float v = hr[k] * scale[k] + shift[k];       // BN affine
      v = (v > 0.f) ? v : 0.2f * v;                // leaky_relu(0.2)
      f.s16[j] = f2bf(v);
    }
    afr[kc] = f.v;
  }

  int srcs[8];
#pragma unroll
  for (int j = 0; j < 8; ++j) srcs[j] = src[e0 + j + (khalf << 3)];

  const int n_lane = lane & 15;
#pragma unroll
  for (int nt = 0; nt < 4; ++nt) {
    v8f acc = {0.f, 0.f, 0.f, 0.f, 0.f, 0.f, 0.f, 0.f};
#pragma unroll
    for (int kc = 0; kc < 2; ++kc) {
      v16bf b = load_bfragT(w3bf, C, nt * 16, kc);
      acc = __builtin_amdgcn_wmma_f32_16x16x32_bf16(false, afr[kc], false, b,
                                                    (short)0, acc, false, false);
    }
    const int n = nt * 16 + n_lane;
    const float bias = b3[n];
#pragma unroll
    for (int j = 0; j < 8; ++j) {
      const int mr = j + (khalf << 3);
      float v = acc[j] + bias;
      v = (v > 0.f) ? v : 0.f;                     // relu
      ew[(size_t)(e0 + mr) * C + n] = v;
      atomicAdd(&deg[(size_t)srcs[j] * C + n], v); // deg = segsum(ew, src)
    }
  }
}

__global__ void dinv_dis_kernel(float* __restrict__ dinv,  // in: deg, out: 1/deg
                                float* __restrict__ dis) { // out: deg^-0.5
  size_t i = (size_t)blockIdx.x * blockDim.x + threadIdx.x;
  if (i < (size_t)N_NODES * C) {
    float d = dinv[i];
    dinv[i] = (d == 0.f) ? 0.f : 1.f / d;
    dis[i]  = (d == 0.f) ? 0.f : rsqrtf(d);
  }
}

// ---------- edge messages: msg_s = segsum(ws@conv_we^T,dst), msg_a = segsum(wa@sym_we^T,dst) ----

__global__ __launch_bounds__(32)
void msg_kernel(const float* __restrict__ ew,
                const float* __restrict__ dinv, const float* __restrict__ dis,
                const int* __restrict__ src, const int* __restrict__ dst,
                const unsigned short* __restrict__ cwebf,
                const unsigned short* __restrict__ swebf,
                float* __restrict__ msg_s, float* __restrict__ msg_a) {
  const int e0 = blockIdx.x * 16;
  const int lane = threadIdx.x & 31;
  const int m = lane & 15;
  const int khalf = lane >> 4;
  const int e = e0 + m;
  const int s = src[e], d = dst[e];
  const int sdd = src[d];  // faithful: wa[dst] indexes edge array by node id
  const float* __restrict__ ew_e = ew + (size_t)e * C;
  const float* __restrict__ ew_d = ew + (size_t)d * C;  // ew[dst[e]] (node id as edge row)
  const float* __restrict__ ew_s = ew + (size_t)s * C;  // ew[src[e]] (node id as edge row)
  const float* __restrict__ di_s = dinv + (size_t)s * C;
  const float* __restrict__ di_x = dinv + (size_t)sdd * C;
  const float* __restrict__ ds_s = dis + (size_t)s * C;
  const float* __restrict__ ds_d = dis + (size_t)d * C;

  v16bf wa_fr[2], ws_fr[2];
#pragma unroll
  for (int kc = 0; kc < 2; ++kc) {
    union { unsigned short s16[16]; v16bf v; } fa, fs;
#pragma unroll
    for (int j = 0; j < 16; ++j) {
      int k = kc * 32 + frag_k(j, khalf);
      float wa  = di_s[k] * ew_e[k] - di_x[k] * ew_d[k];
      float wsv = ds_s[k] * 0.5f * (ew_s[k] + ew_d[k]) * ds_d[k];
      fa.s16[j] = f2bf(wa);
      fs.s16[j] = f2bf(wsv);
    }
    wa_fr[kc] = fa.v;
    ws_fr[kc] = fs.v;
  }

  int dsts[8];
#pragma unroll
  for (int j = 0; j < 8; ++j) dsts[j] = dst[e0 + j + (khalf << 3)];

  const int n_lane = lane & 15;
#pragma unroll
  for (int nt = 0; nt < 4; ++nt) {
    v8f accS = {0.f, 0.f, 0.f, 0.f, 0.f, 0.f, 0.f, 0.f};
    v8f accA = {0.f, 0.f, 0.f, 0.f, 0.f, 0.f, 0.f, 0.f};
#pragma unroll
    for (int kc = 0; kc < 2; ++kc) {
      v16bf bS = load_bfragT(cwebf, C, nt * 16, kc);
      v16bf bA = load_bfragT(swebf, C, nt * 16, kc);
      accS = __builtin_amdgcn_wmma_f32_16x16x32_bf16(false, ws_fr[kc], false, bS,
                                                     (short)0, accS, false, false);
      accA = __builtin_amdgcn_wmma_f32_16x16x32_bf16(false, wa_fr[kc], false, bA,
                                                     (short)0, accA, false, false);
    }
    const int n = nt * 16 + n_lane;
#pragma unroll
    for (int j = 0; j < 8; ++j) {
      atomicAdd(&msg_s[(size_t)dsts[j] * C + n], accS[j]);
      atomicAdd(&msg_a[(size_t)dsts[j] * C + n], accA[j]);
    }
  }
}

// ---------- iteration: y = x@Wa^T, z = x@Ws^T ----------

__global__ __launch_bounds__(32)
void node_gemm_kernel(const unsigned short* __restrict__ x0bf,
                      const unsigned short* __restrict__ Wabf,
                      const unsigned short* __restrict__ Wsbf,
                      float* __restrict__ y, float* __restrict__ z) {
  const int i0 = blockIdx.x * 16;
  const int lane = threadIdx.x & 31;
  const int m = lane & 15;
  const int khalf = lane >> 4;
  const unsigned short* __restrict__ xr = x0bf + (size_t)(i0 + m) * C;

  v16bf afr[2];
#pragma unroll
  for (int kc = 0; kc < 2; ++kc)
    afr[kc] = load_frag_pair(xr + kc * 32 + (khalf << 3));

  const int n_lane = lane & 15;
#pragma unroll
  for (int nt = 0; nt < 4; ++nt) {
    v8f accY = {0.f, 0.f, 0.f, 0.f, 0.f, 0.f, 0.f, 0.f};
    v8f accZ = {0.f, 0.f, 0.f, 0.f, 0.f, 0.f, 0.f, 0.f};
#pragma unroll
    for (int kc = 0; kc < 2; ++kc) {
      v16bf bY = load_bfragT(Wabf, C, nt * 16, kc);
      v16bf bZ = load_bfragT(Wsbf, C, nt * 16, kc);
      accY = __builtin_amdgcn_wmma_f32_16x16x32_bf16(false, afr[kc], false, bY,
                                                     (short)0, accY, false, false);
      accZ = __builtin_amdgcn_wmma_f32_16x16x32_bf16(false, afr[kc], false, bZ,
                                                     (short)0, accZ, false, false);
    }
    const int n = nt * 16 + n_lane;
#pragma unroll
    for (int j = 0; j < 8; ++j) {
      const int mr = j + (khalf << 3);
      y[(size_t)(i0 + mr) * C + n] = accY[j];
      z[(size_t)(i0 + mr) * C + n] = accZ[j];
    }
  }
}

__global__ void edge_scatter_kernel(const float* __restrict__ y, const float* __restrict__ z,
                                    const int* __restrict__ src, const int* __restrict__ dst,
                                    float* __restrict__ acc_y, float* __restrict__ acc_z) {
  size_t t = (size_t)blockIdx.x * blockDim.x + threadIdx.x;
  if (t < (size_t)N_EDGES * C) {
    const int c = (int)(t & (size_t)(C - 1));
    const size_t e = t >> 6;
    const int s = src[e], d = dst[e];
    atomicAdd(&acc_y[(size_t)d * C + c], y[(size_t)s * C + c]);
    atomicAdd(&acc_z[(size_t)d * C + c], z[(size_t)s * C + c]);
  }
}

__global__ __launch_bounds__(32)
void update_kernel(float* __restrict__ x,
                   unsigned short* __restrict__ x0bf,
                   const unsigned short* __restrict__ AWbf,
                   const float* __restrict__ acc_y, const float* __restrict__ acc_z,
                   const float* __restrict__ msg_s, const float* __restrict__ msg_a,
                   const float* __restrict__ bias) {
  const int i0 = blockIdx.x * 16;
  const int lane = threadIdx.x & 31;
  const int m = lane & 15;
  const int khalf = lane >> 4;
  const unsigned short* __restrict__ xr = x0bf + (size_t)(i0 + m) * C;

  v16bf afr[2];
#pragma unroll
  for (int kc = 0; kc < 2; ++kc)
    afr[kc] = load_frag_pair(xr + kc * 32 + (khalf << 3));

  const int n_lane = lane & 15;
#pragma unroll
  for (int nt = 0; nt < 4; ++nt) {
    v8f acc = {0.f, 0.f, 0.f, 0.f, 0.f, 0.f, 0.f, 0.f};
#pragma unroll
    for (int kc = 0; kc < 2; ++kc) {
      v16bf b = load_bfragT(AWbf, C, nt * 16, kc);
      acc = __builtin_amdgcn_wmma_f32_16x16x32_bf16(false, afr[kc], false, b,
                                                    (short)0, acc, false, false);
    }
    const int n = nt * 16 + n_lane;
    const float bn = bias[n];
#pragma unroll
    for (int j = 0; j < 8; ++j) {
      const int i = i0 + j + (khalf << 3);
      const size_t idx = (size_t)i * C + n;
      float u = acc[j] + (acc_y[idx] + msg_s[idx]) +
                BETA_C * (acc_z[idx] + msg_a[idx]) + bn;
      float nx = x[idx] + EPS_C * tanhf(u);
      x[idx] = nx;
      x0bf[idx] = f2bf(nx);  // refresh bf16 mirror for next iteration
    }
  }
}

// ---------- readout ----------

__global__ void readout_kernel(const float* __restrict__ x,
                               const float* __restrict__ l1_w, const float* __restrict__ l1_b,
                               const float* __restrict__ l2_w, const float* __restrict__ l2_b,
                               float* __restrict__ out) {
  int i = blockIdx.x * blockDim.x + threadIdx.x;
  if (i >= N_NODES) return;
  float xr[C];
#pragma unroll
  for (int k = 0; k < C; ++k) xr[k] = x[(size_t)i * C + k];
  float hm[C / 2];
#pragma unroll 2
  for (int o = 0; o < C / 2; ++o) {
    float a = l1_b[o];
#pragma unroll
    for (int k = 0; k < C; ++k) a += l1_w[o * C + k] * xr[k];
    hm[o] = (a > 0.f) ? a : 0.01f * a;
  }
#pragma unroll 2
  for (int o = 0; o < OUT_F; ++o) {
    float a = l2_b[o];
#pragma unroll
    for (int k = 0; k < C / 2; ++k) a += l2_w[o * (C / 2) + k] * hm[k];
    out[(size_t)i * OUT_F + o] = (a > 0.f) ? a : 0.01f * a;
  }
}

// ---------- launch ----------

extern "C" void kernel_launch(void* const* d_in, const int* in_sizes, int n_in,
                              void* d_out, int out_size, void* d_ws, size_t ws_size,
                              hipStream_t stream) {
  const float* x_in       = (const float*)d_in[0];
  const int*   ei         = (const int*)d_in[1];
  const float* emb_w      = (const float*)d_in[2];
  const float* emb_b      = (const float*)d_in[3];
  const float* W          = (const float*)d_in[4];
  const float* bias       = (const float*)d_in[5];
  const float* conv_w_raw = (const float*)d_in[6];
  const float* conv_we    = (const float*)d_in[7];
  const float* sym_w_raw  = (const float*)d_in[8];
  const float* sym_we     = (const float*)d_in[9];
  const float* att_w2     = (const float*)d_in[10];
  const float* att_b2     = (const float*)d_in[11];
  const float* att_w3     = (const float*)d_in[12];
  const float* att_b3     = (const float*)d_in[13];
  const float* bn_g       = (const float*)d_in[14];
  const float* bn_b       = (const float*)d_in[15];
  const float* l1_w       = (const float*)d_in[16];
  const float* l1_b       = (const float*)d_in[17];
  const float* l2_w       = (const float*)d_in[18];
  const float* l2_b       = (const float*)d_in[19];

  const int* src = ei;
  const int* dst = ei + N_EDGES;

  const size_t NC = (size_t)N_NODES * C;   // 3.2e6 floats (12.8 MB, L2-resident)
  const size_t EC = (size_t)N_EDGES * C;   // 51.2e6 floats (204.8 MB, streamed)

  float* ws    = (float*)d_ws;
  float* x0    = ws;                 // NC
  float* hbuf  = x0 + NC;            // EC (reused as iter buffers after mlp2)
  float* ewbuf = hbuf + EC;          // EC
  float* dinv  = ewbuf + EC;         // NC (deg, then 1/deg)
  float* dis   = dinv + NC;          // NC
  float* msg_s = dis + NC;           // NC
  float* msg_a = msg_s + NC;         // NC
  float* stat  = msg_a + NC;         // 256: sum|sqs|scale|shift
  unsigned short* w2bf  = (unsigned short*)(stat + 256);  // 8192
  unsigned short* w3bf  = w2bf + 8192;                    // 4096 each below
  unsigned short* cwebf = w3bf + 4096;
  unsigned short* swebf = cwebf + 4096;
  unsigned short* Wabf  = swebf + 4096;
  unsigned short* Wsbf  = Wabf + 4096;
  unsigned short* AWbf  = Wsbf + 4096;
  unsigned short* x0bf  = AWbf + 4096;                    // NC bf16 (6.4 MB mirror)

  // hbuf (EC floats) is dead after mlp2; alias the 4 iteration buffers into it.
  float* acc_y = hbuf;
  float* acc_z = hbuf + NC;
  float* ynode = hbuf + 2 * NC;
  float* znode = hbuf + 3 * NC;

  const int ZB = 256;
  const int nc_blocks = (int)((NC + ZB - 1) / ZB);

  prep_weights_kernel<<<1, 256, 0, stream>>>(W, conv_w_raw, sym_w_raw, conv_we, sym_we,
                                             att_w2, att_w3, w2bf, w3bf, cwebf, swebf,
                                             Wabf, Wsbf, AWbf);
  zero_kernel<<<nc_blocks, ZB, 0, stream>>>(dinv, NC);
  zero_kernel<<<nc_blocks, ZB, 0, stream>>>(msg_s, NC);
  zero_kernel<<<nc_blocks, ZB, 0, stream>>>(msg_a, NC);
  zero_kernel<<<1, 256, 0, stream>>>(stat, 256);

  emb_kernel<<<(N_NODES + 255) / 256, 256, 0, stream>>>(x_in, emb_w, emb_b, x0, x0bf);

  mlp1_kernel<<<N_EDGES / 16, 32, 0, stream>>>(x0bf, src, dst, w2bf, att_b2, hbuf,
                                               stat, stat + 64);
  bn_finalize_kernel<<<1, 64, 0, stream>>>(stat, stat + 64, bn_g, bn_b,
                                           stat + 128, stat + 192);
  mlp2_kernel<<<N_EDGES / 16, 32, 0, stream>>>(hbuf, src, w3bf, att_b3,
                                               stat + 128, stat + 192, ewbuf, dinv);
  dinv_dis_kernel<<<nc_blocks, ZB, 0, stream>>>(dinv, dis);
  msg_kernel<<<N_EDGES / 16, 32, 0, stream>>>(ewbuf, dinv, dis, src, dst,
                                              cwebf, swebf, msg_s, msg_a);

  const int ec_blocks = (int)((EC + ZB - 1) / ZB);
  for (int it = 0; it < N_ITERS; ++it) {
    zero_kernel<<<nc_blocks, ZB, 0, stream>>>(acc_y, NC);
    zero_kernel<<<nc_blocks, ZB, 0, stream>>>(acc_z, NC);
    node_gemm_kernel<<<N_NODES / 16, 32, 0, stream>>>(x0bf, Wabf, Wsbf, ynode, znode);
    edge_scatter_kernel<<<ec_blocks, ZB, 0, stream>>>(ynode, znode, src, dst, acc_y, acc_z);
    update_kernel<<<N_NODES / 16, 32, 0, stream>>>(x0, x0bf, AWbf, acc_y, acc_z,
                                                   msg_s, msg_a, bias);
  }

  readout_kernel<<<(N_NODES + 255) / 256, 256, 0, stream>>>(x0, l1_w, l1_b,
                                                            l2_w, l2_b, (float*)d_out);
}
